// MgSmmSModel_85220741088115
// MI455X (gfx1250) — matrix operand order repaired and checked by reading, not compiled, not run on hardware
//
#include <hip/hip_runtime.h>

// ---------------- CDNA5 WMMA types ----------------
typedef __attribute__((ext_vector_type(16))) __bf16 v16bf;
typedef __attribute__((ext_vector_type(8)))  float  v8f;

constexpr int kB   = 128;   // batch (GEMM M)
constexpr int kT   = 512;   // time steps
constexpr int kH   = 1024;  // hidden (GEMM N and K)
constexpr int kG   = 512;   // gate size
constexpr int kNWG = 16;    // persistent workgroups
constexpr int kTN  = 64;    // N columns per workgroup (16*64 = 1024)

// ---- workspace layout (bytes) ----
constexpr size_t WS_WB  = 0;                          // W_hc bf16 [H][H]        2 MB
constexpr size_t WS_H0  = WS_WB + (size_t)kH*kH*2;    // h ping  bf16 [B][H]   256 KB
constexpr size_t WS_H1  = WS_H0 + (size_t)kB*kH*2;    // h pong  bf16 [B][H]   256 KB
constexpr size_t WS_HT  = WS_H1 + (size_t)kB*kH*2;    // hT f32  [B][H]        512 KB
constexpr size_t WS_V   = WS_HT + (size_t)kB*kH*4;    // v[k]      4 KB
constexpr size_t WS_CT  = WS_V  + (size_t)kH*4;       // cterm[k]  4 KB
constexpr size_t WS_WX  = WS_CT + (size_t)kH*4;       // wxterm[k] 4 KB
constexpr size_t WS_BC  = WS_WX + (size_t)kH*4;       // biasc[n]  4 KB
constexpr size_t WS_CNT = WS_BC + (size_t)kH*4;       // barrier counter

// ---------------------------------------------------------------------------
// prep: fold the whole output head into per-k vectors.
// y[b] = sum_k hT[b,k]*v[k] + sum_k cterm[k] + x[b,T-1]*sum_k wxterm[k] + b1d
// ---------------------------------------------------------------------------
__global__ void prep_kernel(const float* __restrict__ W_h, const float* __restrict__ W_g,
                            const float* __restrict__ W_x, const float* __restrict__ W1d,
                            const float* __restrict__ b_h, const float* __restrict__ b_g,
                            const float* __restrict__ b_x, const float* __restrict__ b_ic,
                            const float* __restrict__ b_hc, const float* __restrict__ b_c,
                            float* __restrict__ v, float* __restrict__ cterm,
                            float* __restrict__ wxterm, float* __restrict__ biasc) {
  int i = blockIdx.x * blockDim.x + threadIdx.x;   // 0..H-1
  if (i >= kH) return;
  float vi = 0.f;
  for (int h = 0; h < kH; ++h) vi += W1d[h] * W_h[(size_t)h * kH + i];
  v[i] = vi;
  float sg = 0.f;
  for (int j = 0; j < kG; ++j) sg += W_g[(size_t)i * kG + j];
  cterm[i]  = W1d[i] * (b_h[i] + sg + b_g[i] + b_x[i]);
  wxterm[i] = W1d[i] * W_x[i];
  biasc[i]  = b_ic[i] + b_hc[i] + b_c[i];
}

// ---------------------------------------------------------------------------
// cvt_init: W_hc f32->bf16, h0 = bf16(x[:,0]*W_ic + biasc), zero barrier ctr.
// ---------------------------------------------------------------------------
__global__ void cvt_init_kernel(const float* __restrict__ W_hc, const float* __restrict__ x,
                                const float* __restrict__ W_ic, const float* __restrict__ biasc,
                                __bf16* __restrict__ Wb, __bf16* __restrict__ h0,
                                int* __restrict__ counter) {
  size_t idx = (size_t)blockIdx.x * blockDim.x + threadIdx.x;
  if (idx == 0) *counter = 0;
  if (idx < (size_t)kH * kH) {
    Wb[idx] = (__bf16)W_hc[idx];
  } else {
    size_t i2 = idx - (size_t)kH * kH;
    if (i2 < (size_t)kB * kH) {
      int b = (int)(i2 / kH), n = (int)(i2 % kH);
      h0[i2] = (__bf16)(x[(size_t)b * kT] * W_ic[n] + biasc[n]);
    }
  }
}

// ---------------------------------------------------------------------------
// grid-wide split barrier: monotonic device-scope counter
// ---------------------------------------------------------------------------
__device__ __forceinline__ void grid_barrier(int* counter, int target, int tid) {
  __syncthreads();
  if (tid == 0) {
    __threadfence();
    __hip_atomic_fetch_add(counter, 1, __ATOMIC_ACQ_REL, __HIP_MEMORY_SCOPE_AGENT);
    while (__hip_atomic_load(counter, __ATOMIC_ACQUIRE, __HIP_MEMORY_SCOPE_AGENT) < target)
      __builtin_amdgcn_s_sleep(2);
    __threadfence();
  }
  __syncthreads();
}

// ---------------------------------------------------------------------------
// one scan step: hnext = bf16( hprev @ W^T + x[:,t]*W_ic + biasc )
// LDS tile rows are PERMUTED (W row n=j*16+l16 stored at LDS row l16*4+j), so
// each lane's 4 B fragments sit at base + j*2048 + k0*2: one base VGPR,
// all ds_load_b128 use immediate offsets (max 8 KB << 64 KB field).
// ---------------------------------------------------------------------------
__device__ __forceinline__ void scan_step(
    int t, bool last, int wg, int wave, int half, int l16,
    const float* __restrict__ x,
    const __bf16* __restrict__ hprev, __bf16* __restrict__ hnext,
    float* __restrict__ hT, const char* bbase,
    const float* wic, const float* bcv) {
  const int mbase = wave * 16;
  v8f acc[4] = { v8f{}, v8f{}, v8f{}, v8f{} };

  const __bf16* arow = hprev + (size_t)(mbase + l16) * kH + half * 8;

#pragma unroll
  for (int k0 = 0; k0 < kH; k0 += 32) {
    // A fragment 16x32 bf16: elems 0-7 at K=k0+8*half, 8-15 at K=k0+16+8*half
    v16bf a;
    ((uint4*)&a)[0] = *(const uint4*)(arow + k0);
    ((uint4*)&a)[1] = *(const uint4*)(arow + k0 + 16);
    // 4 B fragments: single base + immediate offsets (permuted LDS layout)
    v16bf bb[4];
#pragma unroll
    for (int j = 0; j < 4; ++j) {
      const char* brow = bbase + j * 2048 + k0 * 2;
      ((uint4*)&bb[j])[0] = ((const uint4*)brow)[0];
      ((uint4*)&bb[j])[1] = ((const uint4*)brow)[2];
    }
#pragma unroll
    for (int j = 0; j < 4; ++j)
      acc[j] = __builtin_amdgcn_wmma_f32_16x16x32_bf16(
          false, a, false, bb[j], (short)0, acc[j], false, false);
  }

  // epilogue: + x[b,t]*W_ic[n] + biasc[n]
  float xt[8];
#pragma unroll
  for (int e = 0; e < 8; ++e)
    xt[e] = x[(size_t)(mbase + 8 * half + e) * kT + t];
  if (t + 1 < kT)  // global_prefetch_b8 of next step's x column
    __builtin_prefetch(&x[(size_t)(mbase + 8 * half) * kT + t + 1], 0, 1);

#pragma unroll
  for (int j = 0; j < 4; ++j) {
    const int ng = wg * kTN + j * 16 + l16;
#pragma unroll
    for (int e = 0; e < 8; ++e) {
      const int b = mbase + 8 * half + e;        // C layout: M = e + 8*half
      const float f = acc[j][e] + xt[e] * wic[j] + bcv[j];
      hnext[(size_t)b * kH + ng] = (__bf16)f;
      if (last) hT[(size_t)b * kH + ng] = f;
    }
  }
}

// ---------------------------------------------------------------------------
// Persistent bf16-WMMA scan: 511 sequential steps, W tile resident in LDS.
// ---------------------------------------------------------------------------
__global__ void scan_kernel(const float* __restrict__ x, const float* __restrict__ W_ic,
                            const float* __restrict__ biasc, const __bf16* __restrict__ Wb,
                            __bf16* __restrict__ hbuf0, __bf16* __restrict__ hbuf1,
                            float* __restrict__ hT, int* __restrict__ counter) {
  extern __shared__ __align__(16) char smem[];           // [kTN][kH] bf16 = 128 KB
  const int wg   = blockIdx.x;
  const int tid  = threadIdx.x;
  const int wave = tid >> 5;        // wave32
  const int lane = tid & 31;
  const int half = lane >> 4;
  const int l16  = lane & 15;

  // ---- stage W tile into LDS via gfx1250 async copy (ASYNCcnt), applying ----
  // ---- the row permutation: source row n -> LDS row (n%16)*4 + n/16      ----
  {
    const unsigned ldsbase = __builtin_amdgcn_groupstaticsize();
    const char* src = (const char*)(Wb + (size_t)wg * kTN * kH);
    for (int i = tid * 16; i < kTN * kH * 2; i += (int)blockDim.x * 16) {
      const int n   = i >> 11;            // source row (2048 B per row)
      const int col = i & 2047;
      const int p   = ((n & 15) << 2) + (n >> 4);
      const unsigned loff = ldsbase + (unsigned)(p * 2048 + col);
      const char* g = src + i;
      asm volatile("global_load_async_to_lds_b128 %0, %1, off"
                   :: "v"(loff), "v"(g)
                   : "memory");
    }
    asm volatile("s_wait_asynccnt 0" ::: "memory");
  }
  __syncthreads();

  // per-lane B base: LDS rows l16*4 + j, K half-select adds 16 B
  const char* bbase = smem + (size_t)l16 * 4 * 2048 + half * 16;

  // loop-invariant epilogue constants for this lane
  float wic[4], bcv[4];
#pragma unroll
  for (int j = 0; j < 4; ++j) {
    const int ng = wg * kTN + j * 16 + l16;
    wic[j] = W_ic[ng];
    bcv[j] = biasc[ng];
  }

  // ping-pong with FIXED pointer roles (keeps loads on the GLOBAL pipe)
  int t = 1;
  for (; t + 1 < kT; t += 2) {
    scan_step(t,     false, wg, wave, half, l16, x, hbuf0, hbuf1, hT, bbase, wic, bcv);
    grid_barrier(counter, kNWG * t, tid);
    scan_step(t + 1, false, wg, wave, half, l16, x, hbuf1, hbuf0, hT, bbase, wic, bcv);
    grid_barrier(counter, kNWG * (t + 1), tid);
  }
  // final step (t = kT-1 = 511, reads hbuf0, also writes f32 hT)
  scan_step(t, true, wg, wave, half, l16, x, hbuf0, hbuf1, hT, bbase, wic, bcv);
}

// ---------------------------------------------------------------------------
// out: y[b] = sum_k ( hT[b,k]*v[k] + cterm[k] + x_last*wxterm[k] ) + b1d
// ---------------------------------------------------------------------------
__global__ void out_kernel(const float* __restrict__ hT, const float* __restrict__ v,
                           const float* __restrict__ cterm, const float* __restrict__ wxterm,
                           const float* __restrict__ x, const float* __restrict__ b1d,
                           float* __restrict__ y) {
  __shared__ float red[256];
  const int b = blockIdx.x, tid = threadIdx.x;
  const float xl = x[(size_t)b * kT + (kT - 1)];
  float s = 0.f;
  for (int k = tid; k < kH; k += 256)
    s += hT[(size_t)b * kH + k] * v[k] + cterm[k] + xl * wxterm[k];
  red[tid] = s;
  __syncthreads();
  for (int off = 128; off > 0; off >>= 1) {
    if (tid < off) red[tid] += red[tid + off];
    __syncthreads();
  }
  if (tid == 0) y[b] = red[0] + b1d[0];
}

// ---------------------------------------------------------------------------
extern "C" void kernel_launch(void* const* d_in, const int* in_sizes, int n_in,
                              void* d_out, int out_size, void* d_ws, size_t ws_size,
                              hipStream_t stream) {
  const float* x    = (const float*)d_in[0];
  const float* W_ic = (const float*)d_in[1];
  const float* b_ic = (const float*)d_in[2];
  const float* W_hc = (const float*)d_in[3];
  const float* b_hc = (const float*)d_in[4];
  const float* b_c  = (const float*)d_in[5];
  const float* W_h  = (const float*)d_in[6];
  const float* b_h  = (const float*)d_in[7];
  const float* W_g  = (const float*)d_in[8];
  const float* b_g  = (const float*)d_in[9];
  const float* W_x  = (const float*)d_in[10];
  const float* b_x  = (const float*)d_in[11];
  const float* W_1d = (const float*)d_in[12];
  const float* b_1d = (const float*)d_in[13];

  char* ws = (char*)d_ws;
  __bf16* Wb  = (__bf16*)(ws + WS_WB);
  __bf16* h0  = (__bf16*)(ws + WS_H0);
  __bf16* h1  = (__bf16*)(ws + WS_H1);
  float*  hT  = (float*)(ws + WS_HT);
  float*  v   = (float*)(ws + WS_V);
  float*  ct  = (float*)(ws + WS_CT);
  float*  wx  = (float*)(ws + WS_WX);
  float*  bc  = (float*)(ws + WS_BC);
  int*    cnt = (int*)(ws + WS_CNT);

  prep_kernel<<<4, 256, 0, stream>>>(W_h, W_g, W_x, W_1d, b_h, b_g, b_x,
                                     b_ic, b_hc, b_c, v, ct, wx, bc);
  cvt_init_kernel<<<(kH * kH + kB * kH) / 256, 256, 0, stream>>>(
      W_hc, x, W_ic, bc, Wb, h0, cnt);
  scan_kernel<<<kNWG, 256, kTN * kH * 2, stream>>>(x, W_ic, bc, Wb, h0, h1, hT, cnt);
  out_kernel<<<kB, 256, 0, stream>>>(hT, v, ct, wx, x, b_1d, (float*)d_out);
}
